// localTransformer_79577154060994
// MI455X (gfx1250) — matrix operand order repaired and checked by reading, not compiled, not run on hardware
//
#include <hip/hip_runtime.h>

typedef __bf16 bf16;
typedef __attribute__((ext_vector_type(16))) __bf16 v16bf;
typedef __attribute__((ext_vector_type(8)))  float   v8f;

__device__ __forceinline__ float gelu_exact(float z) {
  return 0.5f * z * (1.0f + erff(z * 0.70710678118654752f));
}

// ---------------------------------------------------------------------------
// Weight pre-pack: convert f32 weights to bf16 WMMA A-fragment layout.
// A packing used here (contiguous k, not AMD's table order): M=lane%16, K in {16h+0..16h+15} with
// h=lane/16; element i -> K = 16h+i (matches contiguous B loads).
// Fragment (mt,kt) stored contiguously: [mt][kt][lane][16].
// Also builds B1eff[c][j] = b1[j] + sum_g bv[g*16+c] * W1[g,j]  (bv folded).
// ---------------------------------------------------------------------------
__global__ void lt_prepack(const float* __restrict__ Wv, const float* __restrict__ Wd,
                           const float* __restrict__ W1, const float* __restrict__ b1,
                           const float* __restrict__ W2, const float* __restrict__ bv,
                           bf16* __restrict__ wvF, bf16* __restrict__ wdF,
                           bf16* __restrict__ w1F, bf16* __restrict__ w2F,
                           float* __restrict__ b1e)
{
  const int i = blockIdx.x * blockDim.x + threadIdx.x;
  if (i < 65536) {                 // Wv: 16 mt x 8 kt x 32 lane x 16
    int e = i & 15, lane = (i >> 4) & 31, kt = (i >> 9) & 7, mt = i >> 12;
    int hf = lane >> 4, m = mt * 16 + (lane & 15);
    int kk = 16 * hf + e;
    wvF[i] = (bf16)Wv[m * 256 + kt * 32 + kk];
  }
  if (i < 131072) {                // Wd: 16 mt x 16 kt x 32 lane x 16 (K=512)
    int e = i & 15, lane = (i >> 4) & 31, kt = (i >> 9) & 15, mt = i >> 13;
    int hf = lane >> 4, m = mt * 16 + (lane & 15);
    int kk = 16 * hf + e;
    wdF[i] = (bf16)Wd[m * 512 + kt * 32 + kk];
  }
  if (i < 512) {                   // W1^T, W2^T padded to K=32 with zeros
    int e = i & 15, lane = i >> 4;
    int hf = lane >> 4, m = lane & 15;
    int kk = 16 * hf + e;
    w1F[i] = (kk < 16) ? (bf16)W1[kk * 16 + m] : (bf16)0.0f;
    w2F[i] = (kk < 16) ? (bf16)W2[kk * 16 + m] : (bf16)0.0f;
  }
  if (i < 256) {                   // effective layer-1 bias (bv + b1 folded)
    int c = i >> 4, j = i & 15;
    float s = b1[j];
    for (int g = 0; g < 16; ++g) s += bv[g * 16 + c] * W1[g * 16 + j];
    b1e[i] = s;
  }
}

// ---------------------------------------------------------------------------
// Fused pipeline. 256 threads = 8 waves; each workgroup = 128 positions of
// one batch image; each wave owns 16 positions end-to-end.
//   x0 -> (v = Wv*x) -> MLP over g -> x1 -> (v = Wv*x1) -> MLP -> x2
//   out = Wd * [x0 ; x2] + bd
// All GEMMs use v_wmma_f32_16x16x32_bf16 with f32 accumulation.
// Outer loops kept rolled to bound register pressure (no scratch spills).
// ---------------------------------------------------------------------------
__global__ __launch_bounds__(256) void lt_fused(
    const float* __restrict__ x,   const bf16* __restrict__ wvF,
    const bf16* __restrict__ wdF,  const bf16* __restrict__ w1F,
    const bf16* __restrict__ w2F,  const float* __restrict__ b1e,
    const float* __restrict__ b2,  const float* __restrict__ bd,
    float* __restrict__ out)
{
  extern __shared__ char smem[];
  bf16*  x0  = (bf16*)smem;            // [128][256]  bf16 x0 tile
  bf16*  xcb = x0 + 128 * 256;         // [128][256]  current x tile
  bf16*  vb  = xcb + 128 * 256;        // [128][16][16]  v reordered (p,c,g)
  bf16*  hb  = vb + 128 * 256;         // [8][16][16]    per-wave MLP hidden
  float* b1s = (float*)(hb + 8 * 256); // [256]
  float* b2s = b1s + 256;              // [16]
  float* bds = b2s + 16;               // [256]

  const int tid  = threadIdx.x;
  const int wvid = tid >> 5, lane = tid & 31;
  const int hf   = lane >> 4, l16 = lane & 15;
  const int pb   = wvid * 16;          // this wave's position base in tile

  const size_t L  = 16384;             // H*W
  const int    wg = blockIdx.x;        // 1024 wgs: 8 batches x 128 tiles
  const float* xg = x   + ((size_t)(wg >> 7) * 256) * L + (size_t)(wg & 127) * 128;
  float*       og = out + ((size_t)(wg >> 7) * 256) * L + (size_t)(wg & 127) * 128;

  // ---- stage biases + x0 tile (coalesced f32 reads -> bf16 LDS) ----
  b1s[tid] = b1e[tid];
  if (tid < 16) b2s[tid] = b2[tid];
  bds[tid] = bd[tid];
#pragma unroll 2
  for (int ch = wvid; ch < 256; ch += 8) {
    const float4 f = *(const float4*)(xg + (size_t)ch * L + lane * 4);
    const int p = lane * 4;
    x0[(p + 0) * 256 + ch] = (bf16)f.x;
    x0[(p + 1) * 256 + ch] = (bf16)f.y;
    x0[(p + 2) * 256 + ch] = (bf16)f.z;
    x0[(p + 3) * 256 + ch] = (bf16)f.w;
  }
  __syncthreads();

  const bf16* xin = x0;
#pragma unroll 1
  for (int it = 0; it < 2; ++it) {
    // ---- v = Wv * x  (M=256, K=256, N=16 per wave) ----
    // B fragments stay register-resident across the m-loop (64 VGPRs).
    v16bf Bf[8];
#pragma unroll
    for (int kt = 0; kt < 8; ++kt)
      Bf[kt] = *(const v16bf*)(xin + (pb + l16) * 256 + kt * 32 + hf * 16);

#pragma unroll 1
    for (int mt = 0; mt < 16; ++mt) {   // mt == group index g
      v8f acc = {0.f, 0.f, 0.f, 0.f, 0.f, 0.f, 0.f, 0.f};
#pragma unroll
      for (int kt = 0; kt < 8; ++kt) {
        v16bf Af = *(const v16bf*)(wvF + ((mt * 8 + kt) * 32 + lane) * 16);
        acc = __builtin_amdgcn_wmma_f32_16x16x32_bf16(
            false, Af, false, Bf[kt], (short)0, acc, false, false);
      }
      // scatter v to (p, c, g) so the MLP's K-dim (g) is contiguous
#pragma unroll
      for (int r = 0; r < 8; ++r)
        vb[((pb + l16) * 16 + (r + 8 * hf)) * 16 + mt] = (bf16)acc[r];
    }

    // ---- MLP across g (K padded 16->32 with zero A rows) ----
    v16bf A1 = *(const v16bf*)(w1F + lane * 16);
    v16bf A2 = *(const v16bf*)(w2F + lane * 16);
#pragma unroll 1
    for (int nt = 0; nt < 16; ++nt) {   // nt == c (minor channel index)
      v16bf Bm = *(const v16bf*)(vb + ((pb + l16) * 16 + nt) * 16);
      v8f h = {0.f, 0.f, 0.f, 0.f, 0.f, 0.f, 0.f, 0.f};
      h = __builtin_amdgcn_wmma_f32_16x16x32_bf16(
          false, A1, false, Bm, (short)0, h, false, false);
#pragma unroll
      for (int r = 0; r < 8; ++r) {
        float z = h[r] + b1s[nt * 16 + r + 8 * hf];   // B1eff (bv,b1 folded)
        hb[wvid * 256 + l16 * 16 + r + 8 * hf] = (bf16)gelu_exact(z);
      }
      v16bf Bh = *(const v16bf*)(hb + wvid * 256 + l16 * 16);
      v8f o2 = {0.f, 0.f, 0.f, 0.f, 0.f, 0.f, 0.f, 0.f};
      o2 = __builtin_amdgcn_wmma_f32_16x16x32_bf16(
          false, A2, false, Bh, (short)0, o2, false, false);
#pragma unroll
      for (int r = 0; r < 8; ++r) {
        float z = o2[r] + b2s[r + 8 * hf];
        xcb[(pb + l16) * 256 + (r + 8 * hf) * 16 + nt] = (bf16)z;  // ch=g'*16+c
      }
    }
    xin = xcb;
  }

  // ---- out = Wd * [x0 ; x2] + bd  (M=256, K=512, N=16 per wave) ----
#pragma unroll 1
  for (int mt = 0; mt < 16; ++mt) {
    v8f acc = {0.f, 0.f, 0.f, 0.f, 0.f, 0.f, 0.f, 0.f};
#pragma unroll 4
    for (int kt = 0; kt < 8; ++kt) {    // K 0..255 from x0
      v16bf Bv = *(const v16bf*)(x0 + (pb + l16) * 256 + kt * 32 + hf * 16);
      v16bf Af = *(const v16bf*)(wdF + ((mt * 16 + kt) * 32 + lane) * 16);
      acc = __builtin_amdgcn_wmma_f32_16x16x32_bf16(
          false, Af, false, Bv, (short)0, acc, false, false);
    }
#pragma unroll 4
    for (int kt = 0; kt < 8; ++kt) {    // K 256..511 from x2
      v16bf Bv = *(const v16bf*)(xcb + (pb + l16) * 256 + kt * 32 + hf * 16);
      v16bf Af = *(const v16bf*)(wdF + ((mt * 16 + 8 + kt) * 32 + lane) * 16);
      acc = __builtin_amdgcn_wmma_f32_16x16x32_bf16(
          false, Af, false, Bv, (short)0, acc, false, false);
    }
#pragma unroll
    for (int r = 0; r < 8; ++r) {
      const int o = mt * 16 + r + 8 * hf;
      og[(size_t)o * L + pb + l16] = acc[r] + bds[o];
    }
  }
}

extern "C" void kernel_launch(void* const* d_in, const int* in_sizes, int n_in,
                              void* d_out, int out_size, void* d_ws, size_t ws_size,
                              hipStream_t stream) {
  (void)in_sizes; (void)n_in; (void)out_size; (void)ws_size;
  const float* x  = (const float*)d_in[0];
  const float* Wv = (const float*)d_in[3];
  const float* bv = (const float*)d_in[4];
  const float* W1 = (const float*)d_in[13];
  const float* b1 = (const float*)d_in[14];
  const float* W2 = (const float*)d_in[15];
  const float* b2 = (const float*)d_in[16];
  const float* Wd = (const float*)d_in[17];
  const float* bd = (const float*)d_in[18];

  char* ws = (char*)d_ws;
  bf16*  wvF = (bf16*)(ws);                              // 131072 B
  bf16*  wdF = (bf16*)(ws + 131072);                     // 262144 B
  bf16*  w1F = (bf16*)(ws + 131072 + 262144);            // 1024 B
  bf16*  w2F = (bf16*)(ws + 131072 + 262144 + 1024);     // 1024 B
  float* b1e = (float*)(ws + 131072 + 262144 + 2048);    // 1024 B

  lt_prepack<<<512, 256, 0, stream>>>(Wv, Wd, W1, b1, W2, bv,
                                      wvF, wdF, w1F, w2F, b1e);

  const int SMEM = (128 * 256 * 3 + 8 * 256) * 2 + 256 * 4 + 16 * 4 + 256 * 4;
  lt_fused<<<1024, 256, SMEM, stream>>>(x, wvF, wdF, w1F, w2F, b1e, b2, bd,
                                        (float*)d_out);
}
